// MultimodalQuantumFusion_13554916786225
// MI455X (gfx1250) — matrix operand order repaired and checked by reading, not compiled, not run on hardware
//
#include <hip/hip_runtime.h>
#include <hip/hip_bf16.h>
#include <hip/hip_fp16.h>

typedef __attribute__((ext_vector_type(16))) _Float16 v16h;
typedef __attribute__((ext_vector_type(8)))  _Float16 v8h;
typedef __attribute__((ext_vector_type(8)))  float    v8f;

#define B_SZ   2048
#define L_SZ   2048
#define T_SZ   512
#define NTILE  128        // B_SZ / 16

// ---------------------------------------------------------------------------
// helpers
// ---------------------------------------------------------------------------
__device__ __forceinline__ float fsig(float x) {
    return 1.f / (1.f + __expf(-x));
}
__device__ __forceinline__ float ftanh(float x) {
    x = fminf(fmaxf(x, -15.f), 15.f);
    float e = __expf(2.f * x);
    return (e - 1.f) / (e + 1.f);
}

// A-fragment (16x32 f16, row-major src, stride 32): per ISA layout,
// lane<16 holds row=lane, K = {0..7, 16..23}; lane>=16 holds K = {8..15, 24..31}
__device__ __forceinline__ v16h load_a_frag(const _Float16* base, int lane) {
    int m  = lane & 15;
    int kh = (lane >> 4) << 3;          // 0 or 8
    v8h lo = *(const v8h*)(base + m * 32 + kh);
    v8h hi = *(const v8h*)(base + m * 32 + 16 + kh);
    v16h a;
#pragma unroll
    for (int i = 0; i < 8; ++i) { a[i] = lo[i]; a[i + 8] = hi[i]; }
    return a;
}

// B-fragment (32x16 f16) from K-major [32][128] weight-transpose in LDS:
// N = n0 + (lane&15); lane<16 holds K = {0..7, 16..23}, lane>=16 holds K = {8..15, 24..31}
__device__ __forceinline__ v16h load_b_frag(const _Float16* wt, int n0, int lane) {
    int n  = n0 + (lane & 15);
    int kh = (lane >> 4) << 3;
    v16h b;
#pragma unroll
    for (int i = 0; i < 8; ++i) { b[i] = wt[(kh + i) * 128 + n]; b[i + 8] = wt[(16 + kh + i) * 128 + n]; }
    return b;
}

// ---------------------------------------------------------------------------
// Kernel 1: conv1 + relu + pool2 + conv2 + relu + pool2  -> seq (f16)
// seq layout: [branch][btile][t][16 rows][32 ch]
// ---------------------------------------------------------------------------
__global__ __launch_bounds__(256)
void conv_branch_kernel(const float* __restrict__ x0, const float* __restrict__ x1,
                        const float* __restrict__ x2, const float* __restrict__ x3,
                        const float* __restrict__ c1w, const float* __restrict__ c1b,
                        const float* __restrict__ c2w, const float* __restrict__ c2b,
                        _Float16* __restrict__ seq)
{
    __shared__ float    xs[L_SZ];
    __shared__ _Float16 a1h[16 * 1024];
    __shared__ float    w2s[32 * 16 * 3];
    __shared__ float    w1s[16 * 3];
    __shared__ float    b1s[16];
    __shared__ float    b2s[32];

    const int b = blockIdx.x, br = blockIdx.y, tid = threadIdx.x;
    const float* xin = (br == 0) ? x0 : (br == 1) ? x1 : (br == 2) ? x2 : x3;

    for (int i = tid; i < L_SZ; i += 256) xs[i] = xin[(size_t)b * L_SZ + i];
    for (int i = tid; i < 48;   i += 256) w1s[i] = c1w[br * 48 + i];
    for (int i = tid; i < 16;   i += 256) b1s[i] = c1b[br * 16 + i];
    for (int i = tid; i < 1536; i += 256) w2s[i] = c2w[br * 1536 + i];
    for (int i = tid; i < 32;   i += 256) b2s[i] = c2b[br * 32 + i];
    __syncthreads();

    // conv1 (1->16, k=3, pad=1) + relu + maxpool2
    for (int idx = tid; idx < 16 * 1024; idx += 256) {
        int ch = idx >> 10, pp = idx & 1023;
        float w0 = w1s[ch * 3], w1 = w1s[ch * 3 + 1], w2 = w1s[ch * 3 + 2], bb = b1s[ch];
        float m = 0.f;
#pragma unroll
        for (int d = 0; d < 2; ++d) {
            int p = 2 * pp + d;
            float xl = (p > 0) ? xs[p - 1] : 0.f;
            float xc = xs[p];
            float xr = (p < L_SZ - 1) ? xs[p + 1] : 0.f;
            float v = fmaxf(bb + w0 * xl + w1 * xc + w2 * xr, 0.f);
            m = fmaxf(m, v);
        }
        a1h[ch * 1024 + pp] = (_Float16)m;
    }
    __syncthreads();

    // conv2 (16->32, k=3, pad=1) + relu + maxpool2 -> store f16 tile rows
    size_t base = (((size_t)br * NTILE + (size_t)(b >> 4)) * T_SZ) * (16 * 32)
                + (size_t)(b & 15) * 32;
    for (int idx = tid; idx < 32 * 512; idx += 256) {
        int oc = idx >> 9, t = idx & 511;
        float m = 0.f;
#pragma unroll
        for (int d = 0; d < 2; ++d) {
            int p = 2 * t + d;
            float v = b2s[oc];
            for (int ic = 0; ic < 16; ++ic) {
                const _Float16* row = &a1h[ic * 1024];
                const float*    w   = &w2s[(oc * 16 + ic) * 3];
                float al = (p > 0)    ? (float)row[p - 1] : 0.f;
                float ac = (float)row[p];
                float ar = (p < 1023) ? (float)row[p + 1] : 0.f;
                v += w[0] * al + w[1] * ac + w[2] * ar;
            }
            m = fmaxf(m, fmaxf(v, 0.f));
        }
        seq[base + (size_t)t * (16 * 32) + oc] = (_Float16)m;
    }
}

// ---------------------------------------------------------------------------
// Kernel 2: LSTM over T=512 steps, 4 waves/block, one wave per 16-sample tile.
// Gates via v_wmma_f32_16x16x32_f16 (K=32 matches input/hidden dims exactly).
// All 16 weight B-fragments held in VGPRs (loop-invariant); acc chains start
// from inline-0 C; biases folded into the activation adds.
// ---------------------------------------------------------------------------
__global__ __launch_bounds__(128)
void lstm_kernel(const _Float16* __restrict__ seq,
                 const float* __restrict__ w_ih, const float* __restrict__ w_hh,
                 const float* __restrict__ b_ih, const float* __restrict__ b_hh,
                 float* __restrict__ hN)
{
    __shared__ __align__(16) _Float16 wihT[32 * 128];
    __shared__ __align__(16) _Float16 whhT[32 * 128];
    __shared__ float biasSum[128];
    __shared__ __align__(16) _Float16 hstage[4][16 * 32];

    const int tid  = threadIdx.x;
    const int lane = tid & 31;
    const int wv   = tid >> 5;
    const int br   = blockIdx.y;
    const int tile = blockIdx.x * 4 + wv;

    const float* wi = w_ih + (size_t)br * 128 * 32;
    const float* wh = w_hh + (size_t)br * 128 * 32;
    for (int i = tid; i < 4096; i += 128) {
        int k = i >> 7, n = i & 127;                 // wT[k][n] = W[n][k]
        wihT[i] = (_Float16)wi[n * 32 + k];
        whhT[i] = (_Float16)wh[n * 32 + k];
    }
    for (int n = tid; n < 128; n += 128) biasSum[n] = b_ih[br * 128 + n] + b_hh[br * 128 + n];
    for (int i = lane; i < 512; i += 32) hstage[wv][i] = (_Float16)0.f;
    __syncthreads();

    const int m16  = lane & 15;
    const int roff = (lane >> 4) << 3;

    // loop-invariant weight fragments: tile j covers gate columns 16j..16j+15
    v16h wfi[8], wfh[8];
#pragma unroll
    for (int j = 0; j < 8; ++j) {
        wfi[j] = load_b_frag(wihT, 16 * j, lane);
        wfh[j] = load_b_frag(whhT, 16 * j, lane);
    }
    // loop-invariant biases for this lane's output column (per gate, per jj)
    float bia[2], bfa[2], bga[2], boa[2];
#pragma unroll
    for (int jj = 0; jj < 2; ++jj) {
        bia[jj] = biasSum[      jj * 16 + m16];
        bfa[jj] = biasSum[32  + jj * 16 + m16];
        bga[jj] = biasSum[64  + jj * 16 + m16];
        boa[jj] = biasSum[96  + jj * 16 + m16];
    }

    v16h ah = load_a_frag(hstage[wv], lane);         // zeros
    float cst[2][8];
#pragma unroll
    for (int jj = 0; jj < 2; ++jj)
#pragma unroll
        for (int r = 0; r < 8; ++r) cst[jj][r] = 0.f;

    const _Float16* seqBase = seq + (((size_t)br * NTILE + tile) * T_SZ) * (16 * 32);

    for (int t = 0; t < T_SZ; ++t) {
        v16h ax = load_a_frag(seqBase + (size_t)t * 512, lane);
#pragma unroll
        for (int jj = 0; jj < 2; ++jj) {
            v8f acc_i = {}, acc_f = {}, acc_g = {}, acc_o = {};   // inline-0 C

            acc_i = __builtin_amdgcn_wmma_f32_16x16x32_f16(false, ax, false, wfi[    jj], (short)0, acc_i, false, false);
            acc_i = __builtin_amdgcn_wmma_f32_16x16x32_f16(false, ah, false, wfh[    jj], (short)0, acc_i, false, false);
            acc_f = __builtin_amdgcn_wmma_f32_16x16x32_f16(false, ax, false, wfi[2 + jj], (short)0, acc_f, false, false);
            acc_f = __builtin_amdgcn_wmma_f32_16x16x32_f16(false, ah, false, wfh[2 + jj], (short)0, acc_f, false, false);
            acc_g = __builtin_amdgcn_wmma_f32_16x16x32_f16(false, ax, false, wfi[4 + jj], (short)0, acc_g, false, false);
            acc_g = __builtin_amdgcn_wmma_f32_16x16x32_f16(false, ah, false, wfh[4 + jj], (short)0, acc_g, false, false);
            acc_o = __builtin_amdgcn_wmma_f32_16x16x32_f16(false, ax, false, wfi[6 + jj], (short)0, acc_o, false, false);
            acc_o = __builtin_amdgcn_wmma_f32_16x16x32_f16(false, ah, false, wfh[6 + jj], (short)0, acc_o, false, false);

#pragma unroll
            for (int r = 0; r < 8; ++r) {
                float ig = fsig (acc_i[r] + bia[jj]);
                float fg = fsig (acc_f[r] + bfa[jj]);
                float gg = ftanh(acc_g[r] + bga[jj]);
                float og = fsig (acc_o[r] + boa[jj]);
                float c  = fg * cst[jj][r] + ig * gg;
                cst[jj][r] = c;
                float h  = og * ftanh(c);
                // D layout: row m = r+roff, col = jj*16 + m16
                hstage[wv][(r + roff) * 32 + jj * 16 + m16] = (_Float16)h;
            }
        }
        // LDS ops within a wave are in-order; hstage is per-wave private,
        // so the D->A layout round trip needs no barrier.
        ah = load_a_frag(hstage[wv], lane);
    }

    // write final hidden state h_n: hN[br][sample][32]
    for (int i = lane; i < 512; i += 32) {
        int m = i >> 5, k = i & 31;
        hN[(((size_t)br * B_SZ) + (size_t)tile * 16 + m) * 32 + k] = (float)hstage[wv][i];
    }
}

// ---------------------------------------------------------------------------
// Kernel 3: bottleneck  compressed = tanh(concat(h_n) @ bott_w^T + bott_b)
// ---------------------------------------------------------------------------
__global__ __launch_bounds__(256)
void bottleneck_kernel(const float* __restrict__ hN,
                       const float* __restrict__ bw, const float* __restrict__ bb,
                       float* __restrict__ compressed)
{
    int b = blockIdx.x * 256 + threadIdx.x;
    float acc[8];
#pragma unroll
    for (int j = 0; j < 8; ++j) acc[j] = bb[j];
    for (int br = 0; br < 4; ++br) {
        const float* hp = hN + ((size_t)br * B_SZ + b) * 32;
        for (int k = 0; k < 32; ++k) {
            float v = hp[k];
            int col = br * 32 + k;
#pragma unroll
            for (int j = 0; j < 8; ++j) acc[j] += v * bw[j * 128 + col];
        }
    }
#pragma unroll
    for (int j = 0; j < 8; ++j) compressed[(size_t)b * 8 + j] = ftanh(acc[j]);
}

// ---------------------------------------------------------------------------
// Kernel 4: 8-qubit statevector sim (one wave32 per sample, 8 cplx amps/lane)
// + fused classifier.
// ---------------------------------------------------------------------------
struct cplx { float x, y; };
__device__ __forceinline__ cplx cmul(cplx a, cplx b) { return {a.x*b.x - a.y*b.y, a.x*b.y + a.y*b.x}; }
__device__ __forceinline__ cplx cadd(cplx a, cplx b) { return {a.x + b.x, a.y + b.y}; }
__device__ __forceinline__ cplx cshfl_xor(cplx a, int m) {
    return { __shfl_xor(a.x, m, 32), __shfl_xor(a.y, m, 32) };
}

// bq = flat bit position (7-q); bits >=3 live in the lane index, <3 in the slot
__device__ __forceinline__ void apply1q(cplx amp[8], int bq,
                                        cplx g00, cplx g01, cplx g10, cplx g11, int lane)
{
    if (bq >= 3) {
        int lm = 1 << (bq - 3);
        int mybit = (lane >> (bq - 3)) & 1;
#pragma unroll
        for (int s = 0; s < 8; ++s) {
            cplx part = cshfl_xor(amp[s], lm);
            cplx a0 = mybit ? part : amp[s];
            cplx a1 = mybit ? amp[s] : part;
            cplx r0 = cadd(cmul(g00, a0), cmul(g01, a1));
            cplx r1 = cadd(cmul(g10, a0), cmul(g11, a1));
            amp[s] = mybit ? r1 : r0;
        }
    } else {
        int lm = 1 << bq;
#pragma unroll
        for (int s = 0; s < 8; ++s) {
            if ((s & lm) == 0) {
                cplx a0 = amp[s], a1 = amp[s | lm];
                amp[s]      = cadd(cmul(g00, a0), cmul(g01, a1));
                amp[s | lm] = cadd(cmul(g10, a0), cmul(g11, a1));
            }
        }
    }
}

__device__ __forceinline__ void applyCNOT(cplx amp[8], int bc, int bt, int lane)
{
    if (bt >= 3) {                       // bc > bt >= 3: both bits in lane index
        int lm = 1 << (bt - 3);
        int cb = (lane >> (bc - 3)) & 1;
#pragma unroll
        for (int s = 0; s < 8; ++s) {
            cplx part = cshfl_xor(amp[s], lm);
            if (cb) amp[s] = part;
        }
    } else {
        int tm = 1 << bt;
#pragma unroll
        for (int s = 0; s < 8; ++s) {
            if ((s & tm) == 0) {
                int cb = (bc >= 3) ? ((lane >> (bc - 3)) & 1) : ((s >> bc) & 1);
                if (cb) { cplx t = amp[s]; amp[s] = amp[s | tm]; amp[s | tm] = t; }
            }
        }
    }
}

__global__ __launch_bounds__(256)
void quantum_kernel(const float* __restrict__ compressed, const float* __restrict__ qw,
                    const float* __restrict__ clf_w, const float* __restrict__ clf_b,
                    float* __restrict__ out)
{
    const float PI = 3.14159265358979323846f;
    int lane = threadIdx.x & 31;
    int b    = blockIdx.x * 8 + (threadIdx.x >> 5);

    cplx amp[8];
#pragma unroll
    for (int s = 0; s < 8; ++s) amp[s] = {0.f, 0.f};
    if (lane == 0) amp[0] = {1.f, 0.f};

    // RX(angle) embedding, angle = compressed * pi
    for (int q = 0; q < 8; ++q) {
        float half = compressed[(size_t)b * 8 + q] * (PI * 0.5f);
        float c = __cosf(half), s = __sinf(half);
        apply1q(amp, 7 - q, {c, 0.f}, {0.f, -s}, {0.f, -s}, {c, 0.f}, lane);
    }
    for (int l = 0; l < 3; ++l) {
        for (int q = 0; q < 8; ++q) {
            float phi = qw[(l * 8 + q) * 3 + 0];
            float th  = qw[(l * 8 + q) * 3 + 1];
            float om  = qw[(l * 8 + q) * 3 + 2];
            float ct = __cosf(th * 0.5f), st = __sinf(th * 0.5f);
            float A  = (phi + om) * 0.5f, Bv = (phi - om) * 0.5f;
            float cA = __cosf(A), sA = __sinf(A), cB = __cosf(Bv), sB = __sinf(Bv);
            // Rot = [[ct e^{-iA}, -st e^{iB}],[st e^{-iB}, ct e^{iA}]]
            apply1q(amp, 7 - q,
                    { ct * cA, -ct * sA}, {-st * cB, -st * sB},
                    { st * cB, -st * sB}, { ct * cA,  ct * sA}, lane);
        }
        int stride = l + 1;
        for (int q = 0; q < 8; ++q) {
            int t = q + stride;
            if (t < 8) applyCNOT(amp, 7 - q, 7 - t, lane);
        }
    }

    // <Z_i> then fused classifier
    float z[8];
#pragma unroll
    for (int q = 0; q < 8; ++q) z[q] = 0.f;
#pragma unroll
    for (int s = 0; s < 8; ++s) {
        int i = (lane << 3) | s;
        float p = amp[s].x * amp[s].x + amp[s].y * amp[s].y;
#pragma unroll
        for (int q = 0; q < 8; ++q) z[q] += ((i >> (7 - q)) & 1) ? -p : p;
    }
    for (int off = 1; off < 32; off <<= 1)
#pragma unroll
        for (int q = 0; q < 8; ++q) z[q] += __shfl_xor(z[q], off, 32);

    if (lane < 3) {
        float lg = clf_b[lane];
#pragma unroll
        for (int q = 0; q < 8; ++q) lg += z[q] * clf_w[lane * 8 + q];
        out[(size_t)b * 3 + lane] = lg;
    }
}

// ---------------------------------------------------------------------------
// launch
// ---------------------------------------------------------------------------
extern "C" void kernel_launch(void* const* d_in, const int* in_sizes, int n_in,
                              void* d_out, int out_size, void* d_ws, size_t ws_size,
                              hipStream_t stream)
{
    (void)in_sizes; (void)n_in; (void)out_size; (void)ws_size;

    const float* x0   = (const float*)d_in[0];
    const float* x1   = (const float*)d_in[1];
    const float* x2   = (const float*)d_in[2];
    const float* x3   = (const float*)d_in[3];
    const float* c1w  = (const float*)d_in[4];
    const float* c1b  = (const float*)d_in[5];
    const float* c2w  = (const float*)d_in[6];
    const float* c2b  = (const float*)d_in[7];
    const float* wih  = (const float*)d_in[8];
    const float* whh  = (const float*)d_in[9];
    const float* bih  = (const float*)d_in[10];
    const float* bhh  = (const float*)d_in[11];
    const float* bw   = (const float*)d_in[12];
    const float* bb   = (const float*)d_in[13];
    const float* qw   = (const float*)d_in[14];
    const float* clfw = (const float*)d_in[15];
    const float* clfb = (const float*)d_in[16];
    float* out = (float*)d_out;

    // workspace layout
    _Float16* seq = (_Float16*)d_ws;
    size_t seqElems = (size_t)4 * NTILE * T_SZ * 16 * 32;               // 134M f16 = 256 MB
    float* hN = (float*)((char*)d_ws + seqElems * sizeof(_Float16));    // [4][B][32]
    float* compressed = hN + (size_t)4 * B_SZ * 32;                     // [B][8]

    conv_branch_kernel<<<dim3(B_SZ, 4), 256, 0, stream>>>(x0, x1, x2, x3,
                                                          c1w, c1b, c2w, c2b, seq);
    lstm_kernel<<<dim3(NTILE / 4, 4), 128, 0, stream>>>(seq, wih, whh, bih, bhh, hN);
    bottleneck_kernel<<<dim3(B_SZ / 256), 256, 0, stream>>>(hN, bw, bb, compressed);
    quantum_kernel<<<dim3(B_SZ / 8), 256, 0, stream>>>(compressed, qw, clfw, clfb, out);
}
